// PPO_61460982005895
// MI455X (gfx1250) — matrix operand order repaired and checked by reading, not compiled, not run on hardware
//
#include <hip/hip_runtime.h>

typedef __attribute__((ext_vector_type(16))) __bf16 v16bf;
typedef __attribute__((ext_vector_type(8)))  float  v8f;

#define GAMMA_ 0.98f
#define LMBDA_ 0.95f

constexpr int BB = 1024, TT = 512, OBS = 64, HID = 256, NAC = 8;
constexpr int NROWS = BB * TT;   // 524288

// ---------------------------------------------------------------- helpers

__device__ __forceinline__ float fast_tanh(float x) {
    // tanh(x) = 1 - 2/(exp(2x)+1); exp->inf gives 1, exp->0 gives -1 (correct limits)
    float e = __expf(2.0f * x);
    return 1.0f - 2.0f * __builtin_amdgcn_rcpf(e + 1.0f);
}

// A-matrix (16x32 bf16) register build from a row-major f32 row.
// Lane l<16 : row M=l,     elems j<8 -> K=k0+j,     j>=8 -> K=k0+16+(j-8)
// Lane l>=16: row M=l-16,  elems j<8 -> K=k0+8+j,   j>=8 -> K=k0+24+(j-8)
__device__ __forceinline__ v16bf load_A_f32(const float* __restrict__ row, int k0, int hi8) {
    v16bf a;
    int koff = k0 + hi8;
#pragma unroll
    for (int j = 0; j < 8; ++j) a[j]     = (__bf16)row[koff + j];
#pragma unroll
    for (int j = 0; j < 8; ++j) a[j + 8] = (__bf16)row[koff + 16 + j];
    return a;
}

// B tile pre-packed so each lane reads its 16 bf16 contiguously (2x b128).
__device__ __forceinline__ v16bf load_B_pk(const __bf16* __restrict__ pk, int idx) {
    const __bf16* p = pk + ((size_t)idx << 4);
    v16bf b;
#pragma unroll
    for (int j = 0; j < 16; ++j) b[j] = p[j];
    return b;
}

// A fragment from the LDS policy-hidden buffer (bf16 row-major, padded stride).
__device__ __forceinline__ v16bf load_A_lds(const __bf16* row, int kt, int hi8) {
    v16bf a;
    int koff = kt * 32 + hi8;
#pragma unroll
    for (int j = 0; j < 8; ++j) a[j]     = row[koff + j];
#pragma unroll
    for (int j = 0; j < 8; ++j) a[j + 8] = row[koff + 16 + j];
    return a;
}

#define WMMA_BF16(A, Bt, C) \
    __builtin_amdgcn_wmma_f32_16x16x32_bf16(false, (A), false, (Bt), (short)0, (C), false, false)

// ---------------------------------------------------------------- kernel 0: weight packing
// Packed B layout for K-tile kt, N-tile nt, lane l, elem j:
//   n = nt*16 + (l&15);  k = kt*32 + ((l>=16)?8:0) + (j&7) + ((j>=8)?16:0)
__global__ void pack_weights(const float* __restrict__ W1, const float* __restrict__ V1,
                             const float* __restrict__ W2,
                             __bf16* __restrict__ pW1, __bf16* __restrict__ pV1,
                             __bf16* __restrict__ pW2) {
    int p = blockIdx.x * blockDim.x + threadIdx.x;
    if (p < 16384) {  // 16 n-tiles * 2 k-tiles * 32 lanes * 16 elems
        int j = p & 15, lane = (p >> 4) & 31, kt = (p >> 9) & 1, nt = p >> 10;
        int n = nt * 16 + (lane & 15);
        int k = kt * 32 + ((lane & 16) ? 8 : 0) + (j & 7) + ((j & 8) ? 16 : 0);
        pW1[p] = (__bf16)W1[k * HID + n];
        pV1[p] = (__bf16)V1[k * HID + n];
    }
    if (p < 4096) {   // W2: 8 k-tiles (K=256), N padded 8->16 with zeros
        int j = p & 15, lane = (p >> 4) & 31, kt = p >> 9;
        int n = lane & 15;
        int k = kt * 32 + ((lane & 16) ? 8 : 0) + (j & 7) + ((j & 8) ? 16 : 0);
        pW2[p] = (n < NAC) ? (__bf16)W2[k * NAC + n] : (__bf16)0.0f;
    }
}

// ---------------------------------------------------------------- kernel 1: fused forward
// 4 waves/block, 32 rows (two 16-row M-tiles) per wave: every B-tile load feeds
// 2 M-tiles, and the v(s)/v(s') passes share one pV1 sweep (4x B-reuse there).
__global__ __launch_bounds__(128) void ppo_fused_fwd(
    const float* __restrict__ s, const float* __restrict__ sp,
    const float* __restrict__ rw, const int* __restrict__ act,
    const float* __restrict__ pa, const float* __restrict__ dm,
    const __bf16* __restrict__ pW1, const __bf16* __restrict__ pV1,
    const __bf16* __restrict__ pW2,
    const float* __restrict__ b1, const float* __restrict__ c1,
    const float* __restrict__ b2, const float* __restrict__ V2,
    const float* __restrict__ c2,
    float* __restrict__ deltaO, float* __restrict__ vlO, float* __restrict__ rtO) {

    __shared__ __bf16 hp[4][32][264];   // policy hidden (bf16, padded rows)
    __shared__ float  vbuf[4][2][32];   // v(s), v(s') per row
    __shared__ float  lbuf[4][32][8];   // logits per row

    const int w    = threadIdx.x >> 5;
    const int lane = threadIdx.x & 31;
    const int col0 = lane & 15;
    const int hi8  = (lane & 16) ? 8 : 0;
    const int r0   = hi8;               // rows 0-7 vs 8-15 within a C tile
    const int base = (blockIdx.x * 4 + w) * 32;
    const float* s0 = s  + (size_t)(base + col0) * OBS;        // tile0 A row
    const float* s1 = s  + (size_t)(base + 16 + col0) * OBS;   // tile1 A row
    const float* q0 = sp + (size_t)(base + col0) * OBS;
    const float* q1 = sp + (size_t)(base + 16 + col0) * OBS;

    __builtin_prefetch(q0, 0, 1);       // global_prefetch for the s' stream
    __builtin_prefetch(q1, 0, 1);

    v16bf as0a = load_A_f32(s0, 0, hi8), as0b = load_A_f32(s0, 32, hi8);
    v16bf as1a = load_A_f32(s1, 0, hi8), as1b = load_A_f32(s1, 32, hi8);
    v16bf ap0a = load_A_f32(q0, 0, hi8), ap0b = load_A_f32(q0, 32, hi8);
    v16bf ap1a = load_A_f32(q1, 0, hi8), ap1b = load_A_f32(q1, 32, hi8);

    // ----- value net on s AND s' in one pV1 sweep -----
    float va0[8], va1[8], vp0[8], vp1[8];
#pragma unroll
    for (int r = 0; r < 8; ++r) { va0[r] = va1[r] = vp0[r] = vp1[r] = 0.f; }
    for (int nt = 0; nt < 16; ++nt) {
        v16bf bt0 = load_B_pk(pV1, (nt * 2 + 0) * 32 + lane);
        v16bf bt1 = load_B_pk(pV1, (nt * 2 + 1) * 32 + lane);
        v8f c00 = {}, c01 = {}, c10 = {}, c11 = {};
        c00 = WMMA_BF16(as0a, bt0, c00);  c00 = WMMA_BF16(as0b, bt1, c00);
        c01 = WMMA_BF16(as1a, bt0, c01);  c01 = WMMA_BF16(as1b, bt1, c01);
        c10 = WMMA_BF16(ap0a, bt0, c10);  c10 = WMMA_BF16(ap0b, bt1, c10);
        c11 = WMMA_BF16(ap1a, bt0, c11);  c11 = WMMA_BF16(ap1b, bt1, c11);
        int col = nt * 16 + col0;
        float c1v = c1[col], w2v = V2[col];
#pragma unroll
        for (int r = 0; r < 8; ++r) {
            va0[r] += fast_tanh(c00[r] + c1v) * w2v;
            va1[r] += fast_tanh(c01[r] + c1v) * w2v;
            vp0[r] += fast_tanh(c10[r] + c1v) * w2v;
            vp1[r] += fast_tanh(c11[r] + c1v) * w2v;
        }
    }

    // ----- policy hidden on s -> LDS (tanh'd, bf16), both M-tiles per B load -----
    for (int nt = 0; nt < 16; ++nt) {
        v16bf bt0 = load_B_pk(pW1, (nt * 2 + 0) * 32 + lane);
        v16bf bt1 = load_B_pk(pW1, (nt * 2 + 1) * 32 + lane);
        v8f c0 = {}, c1t = {};
        c0  = WMMA_BF16(as0a, bt0, c0);   c0  = WMMA_BF16(as0b, bt1, c0);
        c1t = WMMA_BF16(as1a, bt0, c1t);  c1t = WMMA_BF16(as1b, bt1, c1t);
        int col = nt * 16 + col0;
        float b1v = b1[col];
#pragma unroll
        for (int r = 0; r < 8; ++r) {
            hp[w][r0 + r][col]      = (__bf16)fast_tanh(c0[r]  + b1v);
            hp[w][16 + r0 + r][col] = (__bf16)fast_tanh(c1t[r] + b1v);
        }
    }

    // cross-lane sum over the 16 columns held within each 16-lane group
#pragma unroll
    for (int m = 1; m < 16; m <<= 1) {
#pragma unroll
        for (int r = 0; r < 8; ++r) {
            va0[r] += __shfl_xor(va0[r], m, 32);
            va1[r] += __shfl_xor(va1[r], m, 32);
            vp0[r] += __shfl_xor(vp0[r], m, 32);
            vp1[r] += __shfl_xor(vp1[r], m, 32);
        }
    }
    if (col0 == 0) {
        float c2v = c2[0];
#pragma unroll
        for (int r = 0; r < 8; ++r) {
            vbuf[w][0][r0 + r]      = va0[r] + c2v;
            vbuf[w][0][16 + r0 + r] = va1[r] + c2v;
            vbuf[w][1][r0 + r]      = vp0[r] + c2v;
            vbuf[w][1][16 + r0 + r] = vp1[r] + c2v;
        }
    }
    __syncthreads();

    // ----- logits = Hp @ W2 (+b2) via WMMA, A read back from LDS -----
    v8f l0 = {}, l1 = {};
    const __bf16* h0 = &hp[w][col0][0];        // tile0: this lane's M row
    const __bf16* h1 = &hp[w][16 + col0][0];   // tile1
    for (int kt = 0; kt < 8; ++kt) {
        v16bf ah0 = load_A_lds(h0, kt, hi8);
        v16bf ah1 = load_A_lds(h1, kt, hi8);
        v16bf bw  = load_B_pk(pW2, kt * 32 + lane);
        l0 = WMMA_BF16(ah0, bw, l0);
        l1 = WMMA_BF16(ah1, bw, l1);
    }
    if (col0 < NAC) {
        float b2v = b2[col0];
#pragma unroll
        for (int r = 0; r < 8; ++r) {
            lbuf[w][r0 + r][col0]      = l0[r] + b2v;
            lbuf[w][16 + r0 + r][col0] = l1[r] + b2v;
        }
    }
    __syncthreads();

    // ----- per-row PPO epilogue (all 32 lanes, one row each) -----
    {
        int row = lane;
        size_t g = (size_t)base + row;
        float vs  = vbuf[w][0][row];
        float vsp = vbuf[w][1][row];
        float mk  = dm[g];
        float td    = rw[g] + GAMMA_ * (vsp * mk);
        float delta = td - vs * mk;
        float d  = vs - td;
        float ad = fabsf(d);
        float vl = (ad < 1.0f) ? 0.5f * d * d : (ad - 0.5f);   // smooth_l1
        float l[8];
#pragma unroll
        for (int j = 0; j < 8; ++j) l[j] = lbuf[w][row][j];
        float mx = l[0];
#pragma unroll
        for (int j = 1; j < 8; ++j) mx = fmaxf(mx, l[j]);
        float se = 0.f;
#pragma unroll
        for (int j = 0; j < 8; ++j) se += __expf(l[j] - mx);
        float lse = mx + __logf(se);
        int ai = act[g];
        float pia   = __expf(lbuf[w][row][ai] - lse);
        float ratio = __expf(__logf(pia + 1e-10f) - __logf(pa[g] + 1e-10f));
        deltaO[g] = delta;
        vlO[g]    = vl;
        rtO[g]    = ratio;
    }
}

// ---------------------------------------------------------------- kernel 2: GAE + partial stats
__global__ __launch_bounds__(256) void gae_kernel(const float* __restrict__ delta,
                                                  const float* __restrict__ dm,
                                                  float* __restrict__ adv,
                                                  float* __restrict__ part) {
    int tid = threadIdx.x;
    int b = blockIdx.x * 256 + tid;            // one env per lane (1024 total)
    const float gl = GAMMA_ * LMBDA_;
    const float* dp = delta + (size_t)b * TT;
    const float* mp = dm    + (size_t)b * TT;
    float* ap       = adv   + (size_t)b * TT;
    float g = 0.f, sm = 0.f, sq = 0.f, sw = 0.f;
    for (int t = TT - 1; t >= 0; --t) {
        float m = mp[t];
        g = dp[t] + gl * g * m;
        ap[t] = g;
        sm += g * m; sq += g * g * m; sw += m;
    }
    __shared__ float sb[256];
    sb[tid] = sm; __syncthreads();
    for (int st = 128; st > 0; st >>= 1) { if (tid < st) sb[tid] += sb[tid + st]; __syncthreads(); }
    if (tid == 0) part[blockIdx.x * 3 + 0] = sb[0];
    __syncthreads();
    sb[tid] = sq; __syncthreads();
    for (int st = 128; st > 0; st >>= 1) { if (tid < st) sb[tid] += sb[tid + st]; __syncthreads(); }
    if (tid == 0) part[blockIdx.x * 3 + 1] = sb[0];
    __syncthreads();
    sb[tid] = sw; __syncthreads();
    for (int st = 128; st > 0; st >>= 1) { if (tid < st) sb[tid] += sb[tid + st]; __syncthreads(); }
    if (tid == 0) part[blockIdx.x * 3 + 2] = sb[0];
}

// ---------------------------------------------------------------- kernel 3: mean/std
__global__ void stats_kernel(const float* __restrict__ part, float* __restrict__ stats) {
    if (threadIdx.x == 0 && blockIdx.x == 0) {
        float sm = 0.f, sq = 0.f, sw = 0.f;
        for (int i = 0; i < 4; ++i) { sm += part[i*3]; sq += part[i*3+1]; sw += part[i*3+2]; }
        float mean = sm / sw;
        float var  = fmaxf(sq / sw - mean * mean, 0.f);
        stats[0] = mean;
        stats[1] = sqrtf(var) + 1e-8f;
        stats[2] = sw;
    }
}

// ---------------------------------------------------------------- kernel 4: loss partials
__global__ __launch_bounds__(256) void loss_kernel(const float* __restrict__ adv,
                                                   const float* __restrict__ rt,
                                                   const float* __restrict__ vl,
                                                   const float* __restrict__ dm,
                                                   const float* __restrict__ stats,
                                                   float* __restrict__ part) {
    float mean = stats[0], den = stats[1];
    float acc = 0.f;
    for (int i = blockIdx.x * 256 + threadIdx.x; i < NROWS; i += 256 * 256) {
        float na = (adv[i] - mean) / den;
        float r  = rt[i];
        float s1 = r * na;
        float s2 = fminf(fmaxf(r, 1.0f - 0.1f), 1.0f + 0.1f) * na;
        float pl = -fminf(s1, s2);
        acc += (pl + 0.5f * vl[i]) * dm[i];
    }
    __shared__ float sb[256];
    int tid = threadIdx.x;
    sb[tid] = acc; __syncthreads();
    for (int st = 128; st > 0; st >>= 1) { if (tid < st) sb[tid] += sb[tid + st]; __syncthreads(); }
    if (tid == 0) part[blockIdx.x] = sb[0];
}

// ---------------------------------------------------------------- kernel 5: final scalar
__global__ __launch_bounds__(256) void final_kernel(const float* __restrict__ part,
                                                    const float* __restrict__ stats,
                                                    float* __restrict__ out) {
    __shared__ float sb[256];
    int tid = threadIdx.x;
    sb[tid] = part[tid]; __syncthreads();
    for (int st = 128; st > 0; st >>= 1) { if (tid < st) sb[tid] += sb[tid + st]; __syncthreads(); }
    if (tid == 0) out[0] = sb[0] / stats[2];
}

// ---------------------------------------------------------------- launch
extern "C" void kernel_launch(void* const* d_in, const int* in_sizes, int n_in,
                              void* d_out, int out_size, void* d_ws, size_t ws_size,
                              hipStream_t stream) {
    const float* s   = (const float*)d_in[0];
    const float* sp  = (const float*)d_in[1];
    const float* rw  = (const float*)d_in[2];
    const int*   act = (const int*)  d_in[3];
    const float* pa  = (const float*)d_in[4];
    const float* dm  = (const float*)d_in[5];
    const float* W1  = (const float*)d_in[6];
    const float* b1  = (const float*)d_in[7];
    const float* W2  = (const float*)d_in[8];
    const float* b2  = (const float*)d_in[9];
    const float* V1  = (const float*)d_in[10];
    const float* c1  = (const float*)d_in[11];
    const float* V2  = (const float*)d_in[12];
    const float* c2  = (const float*)d_in[13];

    char* ws = (char*)d_ws;
    __bf16* pW1  = (__bf16*)(ws);                                  // 32 KB
    __bf16* pV1  = (__bf16*)(ws + 32768);                          // 32 KB
    __bf16* pW2  = (__bf16*)(ws + 65536);                          // 8 KB
    float* delta = (float*)(ws + 73728);                           // 2 MB
    float* vloss = (float*)(ws + 73728 + 1 * 2097152);             // 2 MB
    float* ratio = (float*)(ws + 73728 + 2 * 2097152);             // 2 MB
    float* adv   = (float*)(ws + 73728 + 3 * 2097152);             // 2 MB
    float* part2 = (float*)(ws + 73728 + 4 * 2097152);             // 12 floats
    float* stats = (float*)(ws + 73728 + 4 * 2097152 + 64);        // 3 floats
    float* part4 = (float*)(ws + 73728 + 4 * 2097152 + 128);       // 256 floats

    pack_weights<<<64, 256, 0, stream>>>(W1, V1, W2, pW1, pV1, pW2);
    ppo_fused_fwd<<<NROWS / 128, 128, 0, stream>>>(s, sp, rw, act, pa, dm,
                                                   pW1, pV1, pW2, b1, c1, b2, V2, c2,
                                                   delta, vloss, ratio);
    gae_kernel<<<4, 256, 0, stream>>>(delta, dm, adv, part2);
    stats_kernel<<<1, 1, 0, stream>>>(part2, stats);
    loss_kernel<<<256, 256, 0, stream>>>(adv, ratio, vloss, dm, stats, part4);
    final_kernel<<<1, 256, 0, stream>>>(part4, stats, (float*)d_out);
}
